// RNN_41266045780285
// MI455X (gfx1250) — matrix operand order repaired and checked by reading, not compiled, not run on hardware
//
#include <hip/hip_runtime.h>

typedef __attribute__((ext_vector_type(16))) _Float16 v16h;
typedef __attribute__((ext_vector_type(8)))  _Float16 v8h;
typedef __attribute__((ext_vector_type(4)))  _Float16 v4h;
typedef __attribute__((ext_vector_type(8)))  float    v8f;
typedef __attribute__((ext_vector_type(4)))  float    v4f;   // native vector (NT-builtin compatible)

#define HID  128
#define INP  64
#define OUTD 64
#define BSZ  256
#define SEQL 4096
#define MT   16

union AFrag { v16h v; v8h h[2]; };

// Split workgroup barrier: signal as soon as this wave's LDS writes have landed,
// wait only after the overlappable (global-memory-only) work has been issued.
__device__ __forceinline__ void wg_barrier_signal_after_lds() {
  asm volatile("s_wait_dscnt 0x0" ::: "memory");
  asm volatile("s_barrier_signal -1" ::: "memory");
}
__device__ __forceinline__ void wg_barrier_wait() {
  asm volatile("s_barrier_wait -1" ::: "memory");
}

__global__ __launch_bounds__(256, 1)
void rnn_fused_kernel(const float* __restrict__ x,
                      const float* __restrict__ Wh,
                      const float* __restrict__ bh,
                      const float* __restrict__ Wx,
                      const float* __restrict__ bx,
                      const float* __restrict__ Wy,
                      const float* __restrict__ by,
                      float* __restrict__ out)
{
  // double-buffered hidden state (f16) and input tile (f16)
  __shared__ __align__(16) _Float16 sH[2][MT * HID];
  __shared__ __align__(16) _Float16 sX[2][MT * INP];

  const int tid  = threadIdx.x;
  const int lane = tid & 31;
  const int wv   = tid >> 5;           // wave id 0..7, owns hidden cols [wv*16, wv*16+16)
  const int bm   = blockIdx.x * MT;    // batch tile base row
  const int lrow = lane & 15;
  const bool hi  = lane >= 16;
  const int aoff = hi ? 16 : 0;        // A-fragment K sub-offset per lane half
  const int m0   = hi ? 8 : 0;         // C/D row base per lane half

  const int ncol = wv * 16 + lrow;         // hidden column for B-fragments / C tile
  const int ocol = (wv & 3) * 16 + lrow;   // output column (meaningful for waves 0-3)

  // ---- loop-invariant WMMA B fragments in registers (weights as f16) ----
  // B k order used here (contiguous, same as A): lanes 0-15: N=lane, K=0..15 ; lanes 16-31: N=lane-16, K=16..31
  v16h bWh[4], bWx[2], bWy[4];
  #pragma unroll
  for (int kb = 0; kb < 4; ++kb) {
    const int kbase = kb * 32 + (hi ? 16 : 0);
    #pragma unroll
    for (int j = 0; j < 16; ++j)
      bWh[kb][j] = (_Float16)Wh[(kbase + j) * HID + ncol];
  }
  #pragma unroll
  for (int kb = 0; kb < 2; ++kb) {
    const int kbase = kb * 32 + (hi ? 16 : 0);
    #pragma unroll
    for (int j = 0; j < 16; ++j)
      bWx[kb][j] = (_Float16)Wx[(kbase + j) * HID + ncol];
  }
  #pragma unroll
  for (int kb = 0; kb < 4; ++kb) {
    const int kbase = kb * 32 + (hi ? 16 : 0);
    #pragma unroll
    for (int j = 0; j < 16; ++j)
      bWy[kb][j] = (_Float16)Wy[(kbase + j) * OUTD + ocol];
  }

  const float biasH = bh[ncol] + bx[ncol];  // combined recurrent + input bias
  const float biasO = by[ocol];

  // ---- init: h_{-1} = 0 in sH[0]; x_0 into sX[0] ----
  for (int i = tid; i < MT * HID; i += 256) sH[0][i] = (_Float16)0.f;
  const int xrow = tid >> 4;            // 16 threads per batch row
  const int xc   = (tid & 15) * 4;      // 4 floats each (IN=64)
  {
    const v4f v0 = __builtin_nontemporal_load(
        (const v4f*)(x + ((size_t)(bm + xrow) * SEQL + 0) * INP + xc));
    v4h xh = {(_Float16)v0.x, (_Float16)v0.y, (_Float16)v0.z, (_Float16)v0.w};
    *(v4h*)(&sX[0][xrow * INP + xc]) = xh;
  }
  __syncthreads();

  for (int t = 0; t < SEQL; ++t) {
    const int rb = t & 1, wb = rb ^ 1;

    // software prefetch of x_{t+1} (streamed once: nontemporal)
    v4f xnxt = {0.f, 0.f, 0.f, 0.f};
    if (t + 1 < SEQL)
      xnxt = __builtin_nontemporal_load(
          (const v4f*)(x + ((size_t)(bm + xrow) * SEQL + (t + 1)) * INP + xc));

    // A fragments: h_{t-1} (K=128 -> 4 blocks), x_t (K=64 -> 2 blocks)
    // A k order used here (contiguous, same as B): lanes 0-15 K {kb*32+0..15}; lanes 16-31 K {kb*32+16..31}
    AFrag aH[4], aX[2];
    #pragma unroll
    for (int kb = 0; kb < 4; ++kb) {
      const _Float16* p = &sH[rb][lrow * HID + kb * 32 + aoff];
      aH[kb].h[0] = *(const v8h*)p;
      aH[kb].h[1] = *(const v8h*)(p + 8);
    }
    #pragma unroll
    for (int kb = 0; kb < 2; ++kb) {
      const _Float16* p = &sX[rb][lrow * INP + kb * 32 + aoff];
      aX[kb].h[0] = *(const v8h*)p;
      aX[kb].h[1] = *(const v8h*)(p + 8);
    }

    // two independent accumulation chains for ILP, fp32 accumulate
    v8f accA, accB;
    #pragma unroll
    for (int r = 0; r < 8; ++r) { accA[r] = biasH; accB[r] = 0.f; }
    #pragma unroll
    for (int kb = 0; kb < 2; ++kb)
      accA = __builtin_amdgcn_wmma_f32_16x16x32_f16(false, aX[kb].v, false, bWx[kb],
                                                    (short)0, accA, false, false);
    #pragma unroll
    for (int kb = 0; kb < 4; ++kb)
      accB = __builtin_amdgcn_wmma_f32_16x16x32_f16(false, aH[kb].v, false, bWh[kb],
                                                    (short)0, accB, false, false);

    // output projection of h_{t-1} (independent WMMA chain; fills the
    // WMMA->VALU hazard slots of the recurrent chain). Reuses aH fragments.
    const bool do_out = (t > 0) && (wv < 4);
    v8f oa;
    if (do_out) {
      #pragma unroll
      for (int r = 0; r < 8; ++r) oa[r] = biasO;
      #pragma unroll
      for (int kb = 0; kb < 4; ++kb)
        oa = __builtin_amdgcn_wmma_f32_16x16x32_f16(false, aH[kb].v, false, bWy[kb],
                                                    (short)0, oa, false, false);
    }

    // h_t = relu(biasH + x@Wx + h@Wh)
    #pragma unroll
    for (int r = 0; r < 8; ++r) accA[r] = fmaxf(accA[r] + accB[r], 0.f);

    // write h_t (f16) into the other buffer; C layout: row m0+r, col ncol
    #pragma unroll
    for (int r = 0; r < 8; ++r)
      sH[wb][(m0 + r) * HID + ncol] = (_Float16)accA[r];

    // commit prefetched x_{t+1}
    if (t + 1 < SEQL) {
      v4h xh = {(_Float16)xnxt.x, (_Float16)xnxt.y, (_Float16)xnxt.z, (_Float16)xnxt.w};
      *(v4h*)(&sX[wb][xrow * INP + xc]) = xh;
    }

    // All shared-state writes issued -> signal, overlap global stores with barrier
    wg_barrier_signal_after_lds();

    if (do_out) {
      #pragma unroll
      for (int r = 0; r < 8; ++r)
        __builtin_nontemporal_store(
            oa[r], out + ((size_t)(bm + m0 + r) * SEQL + (t - 1)) * OUTD + ocol);
    }

    wg_barrier_wait();
  }

  // epilogue: out_{S-1} from h_{S-1} stored in sH[((S-1)&1)^1]
  if (wv < 4) {
    const int re = ((SEQL - 1) & 1) ^ 1;
    AFrag aHe[4];
    #pragma unroll
    for (int kb = 0; kb < 4; ++kb) {
      const _Float16* p = &sH[re][lrow * HID + kb * 32 + aoff];
      aHe[kb].h[0] = *(const v8h*)p;
      aHe[kb].h[1] = *(const v8h*)(p + 8);
    }
    v8f oa;
    #pragma unroll
    for (int r = 0; r < 8; ++r) oa[r] = biasO;
    #pragma unroll
    for (int kb = 0; kb < 4; ++kb)
      oa = __builtin_amdgcn_wmma_f32_16x16x32_f16(false, aHe[kb].v, false, bWy[kb],
                                                  (short)0, oa, false, false);
    #pragma unroll
    for (int r = 0; r < 8; ++r)
      __builtin_nontemporal_store(
          oa[r], out + ((size_t)(bm + m0 + r) * SEQL + (SEQL - 1)) * OUTD + ocol);
  }
}

extern "C" void kernel_launch(void* const* d_in, const int* in_sizes, int n_in,
                              void* d_out, int out_size, void* d_ws, size_t ws_size,
                              hipStream_t stream) {
  const float* x  = (const float*)d_in[0];
  const float* Wh = (const float*)d_in[1];
  const float* bh = (const float*)d_in[2];
  const float* Wx = (const float*)d_in[3];
  const float* bx = (const float*)d_in[4];
  const float* Wy = (const float*)d_in[5];
  const float* by = (const float*)d_in[6];
  float* out = (float*)d_out;

  dim3 grid(BSZ / MT);   // 16 independent batch tiles
  dim3 block(256);       // 8 waves (wave32), one N-slice each
  rnn_fused_kernel<<<grid, block, 0, stream>>>(x, Wh, bh, Wx, bx, Wy, by, out);
}
